// Transformer_18262200942773
// MI455X (gfx1250) — matrix operand order repaired and checked by reading, not compiled, not run on hardware
//
#include <hip/hip_runtime.h>

typedef __bf16 bf16;
typedef __attribute__((ext_vector_type(16))) __bf16 v16bf;
typedef __attribute__((ext_vector_type(8)))  __bf16 v8bf;
typedef __attribute__((ext_vector_type(8)))  float  v8f;

#define DIMN   1024
#define HEADS  16
#define HD     64
#define MLPD   4096
#define BATCH  4
#define SEQ    1024
#define ROWS   (BATCH*SEQ)   /* 4096 */
#define NLAYER 4
#define LN_EPS 1e-5f
#define ATT_SCALE 0.03125f   /* dim^-0.5 = 1/32 */

#define BM 128
#define BN 128
#define BK 32
#define LPAD 8               /* LDS row stride 40 elems = 80B (16B aligned) */

enum { EPI_QKV = 0, EPI_GELU = 1, EPI_RES = 2 };

union FragU { v16bf v; v8bf h[2]; };

// A-matrix 16x32 bf16 fragment: lane m=lane&15, half=lane>>4.
// elements 0..7 = row[half*8 .. +7], 8..15 = row[16+half*8 .. +7]
__device__ __forceinline__ v16bf frag_a(const bf16* row, int half) {
  FragU u;
  u.h[0] = *(const v8bf*)(row + half * 8);
  u.h[1] = *(const v8bf*)(row + 16 + half * 8);
  return u.v;
}
// B-matrix 32x16 bf16 fragment (row = Bt[n], contiguous in K):
// elements 0..7 = row[half*8 .. +7], 8..15 = row[16+half*8 .. +7]
__device__ __forceinline__ v16bf frag_b(const bf16* row, int half) {
  FragU u;
  u.h[0] = *(const v8bf*)(row + half * 8);
  u.h[1] = *(const v8bf*)(row + 16 + half * 8);
  return u.v;
}

#define WMMA_BF16(a, b, c) \
  __builtin_amdgcn_wmma_f32_16x16x32_bf16(false, (a), false, (b), (short)0, (c), false, false)

// CDNA5 async global -> LDS (ASYNCcnt-tracked, 16B per lane)
__device__ __forceinline__ void async_b128(unsigned int lds_addr, const void* g) {
  asm volatile("global_load_async_to_lds_b128 %0, %1, off"
               :: "v"(lds_addr), "v"((unsigned long long)(uintptr_t)g)
               : "memory");
}
__device__ __forceinline__ void wait_async0() {
  asm volatile("s_wait_asynccnt 0" ::: "memory");
}
__device__ __forceinline__ unsigned int lds_addr_of(const void* p) {
  return (unsigned int)(uintptr_t)p;  // low 32 bits of generic ptr = LDS offset
}

// ---------------------------------------------------------------------------
// Transpose + convert: W [K,N] f32 (row-major)  ->  Wt [N,K] bf16 (row-major)
// ---------------------------------------------------------------------------
__global__ __launch_bounds__(256) void transpose_bf16_kernel(
    const float* __restrict__ W, bf16* __restrict__ Wt, int K, int N) {
  __shared__ float T[32][33];
  const int kt = blockIdx.y * 32, nt = blockIdx.x * 32;
  const int c = threadIdx.x & 31, r0 = threadIdx.x >> 5;
#pragma unroll
  for (int i = 0; i < 4; i++)
    T[r0 + i * 8][c] = W[(size_t)(kt + r0 + i * 8) * N + nt + c];
  __syncthreads();
#pragma unroll
  for (int i = 0; i < 4; i++)
    Wt[(size_t)(nt + r0 + i * 8) * K + kt + c] = (bf16)T[c][r0 + i * 8];
}

// ---------------------------------------------------------------------------
// LayerNorm: f32 in -> bf16 out (one 1024-wide row per block)
// ---------------------------------------------------------------------------
__global__ __launch_bounds__(256) void ln_kernel(
    const float* __restrict__ Xin, const float* __restrict__ gam,
    const float* __restrict__ bet, bf16* __restrict__ Hout) {
  __shared__ float red[256];
  const int row = blockIdx.x;
  const float* xr = Xin + (size_t)row * DIMN;
  const int t = threadIdx.x;
  float v[4];
  float s = 0.f;
#pragma unroll
  for (int i = 0; i < 4; i++) { v[i] = xr[i * 256 + t]; s += v[i]; }
  red[t] = s;
  __syncthreads();
  for (int st = 128; st > 0; st >>= 1) {
    if (t < st) red[t] += red[t + st];
    __syncthreads();
  }
  const float mean = red[0] * (1.0f / DIMN);
  __syncthreads();
  float s2 = 0.f;
#pragma unroll
  for (int i = 0; i < 4; i++) { float d = v[i] - mean; s2 += d * d; }
  red[t] = s2;
  __syncthreads();
  for (int st = 128; st > 0; st >>= 1) {
    if (t < st) red[t] += red[t + st];
    __syncthreads();
  }
  const float rstd = rsqrtf(red[0] * (1.0f / DIMN) + LN_EPS);
#pragma unroll
  for (int i = 0; i < 4; i++) {
    const int c = i * 256 + t;
    Hout[(size_t)row * DIMN + c] = (bf16)((v[i] - mean) * rstd * gam[c] + bet[c]);
  }
}

// ---------------------------------------------------------------------------
// Tiled bf16 WMMA GEMM with double-buffered async LDS staging.
//   C[M,N] = A[M,K] @ Bt[N,K]^T, fused epilogues:
//   EPI_QKV : split columns into q/k ([B,H,N,64]) and v transposed ([B,H,64,N])
//   EPI_GELU: bias + exact GELU -> bf16
//   EPI_RES : bias + residual accumulate into f32 X (in place)
// ---------------------------------------------------------------------------
template <int EPI>
__global__ __launch_bounds__(256) void gemm_kernel(
    const bf16* __restrict__ A, const bf16* __restrict__ Bt,
    const float* __restrict__ bias, float* __restrict__ X,
    bf16* __restrict__ Obf, bf16* __restrict__ Qo, bf16* __restrict__ Ko,
    bf16* __restrict__ Vo, int N, int K) {
  __shared__ __align__(16) bf16 As[2][BM][BK + LPAD];
  __shared__ __align__(16) bf16 Bs[2][BN][BK + LPAD];
  const int t = threadIdx.x;
  const int m0 = blockIdx.y * BM, n0 = blockIdx.x * BN;
  const int w = t >> 5, lane = t & 31;
  const int wrow = w >> 2, wcol = w & 3;         // 2 x 4 wave grid
  const int mlane = lane & 15, half = lane >> 4;
  const int lr = t >> 2, lc = (t & 3) * 8;       // global->LDS mapping
  v8f acc[4][2] = {};

  const unsigned int ABUF = (unsigned int)sizeof(As[0]);
  const unsigned int BBUF = (unsigned int)sizeof(Bs[0]);
  const unsigned int aA0 = lds_addr_of(&As[0][lr][lc]);
  const unsigned int aA1 = lds_addr_of(&As[0][lr + 64][lc]);
  const unsigned int aB0 = lds_addr_of(&Bs[0][lr][lc]);
  const unsigned int aB1 = lds_addr_of(&Bs[0][lr + 64][lc]);
  const bf16* gA0 = &A[(size_t)(m0 + lr) * K + lc];
  const bf16* gA1 = &A[(size_t)(m0 + lr + 64) * K + lc];
  const bf16* gB0 = &Bt[(size_t)(n0 + lr) * K + lc];
  const bf16* gB1 = &Bt[(size_t)(n0 + lr + 64) * K + lc];

  // preload tile 0 into buffer 0
  async_b128(aA0, gA0);
  async_b128(aA1, gA1);
  async_b128(aB0, gB0);
  async_b128(aB1, gB1);

  int buf = 0;
  for (int k0 = 0; k0 < K; k0 += BK) {
    wait_async0();
    __syncthreads();
    const int nk = k0 + BK;
    if (nk < K) {
      const unsigned int nb = (unsigned int)(buf ^ 1);
      async_b128(aA0 + nb * ABUF, gA0 + nk);
      async_b128(aA1 + nb * ABUF, gA1 + nk);
      async_b128(aB0 + nb * BBUF, gB0 + nk);
      async_b128(aB1 + nb * BBUF, gB1 + nk);
    }
    v16bf bfr[2];
#pragma unroll
    for (int j = 0; j < 2; j++)
      bfr[j] = frag_b(&Bs[buf][wcol * 32 + j * 16 + mlane][0], half);
#pragma unroll
    for (int i = 0; i < 4; i++) {
      v16bf af = frag_a(&As[buf][wrow * 64 + i * 16 + mlane][0], half);
#pragma unroll
      for (int j = 0; j < 2; j++) acc[i][j] = WMMA_BF16(af, bfr[j], acc[i][j]);
    }
    buf ^= 1;
  }

#pragma unroll
  for (int i = 0; i < 4; i++) {
#pragma unroll
    for (int j = 0; j < 2; j++) {
#pragma unroll
      for (int r = 0; r < 8; r++) {
        const float v = acc[i][j][r];
        const int gm = m0 + wrow * 64 + i * 16 + half * 8 + r;
        const int gn = n0 + wcol * 32 + j * 16 + mlane;
        if constexpr (EPI == EPI_RES) {
          X[(size_t)gm * N + gn] += v + bias[gn];
        } else if constexpr (EPI == EPI_GELU) {
          const float u = v + bias[gn];
          Obf[(size_t)gm * N + gn] =
              (bf16)(0.5f * u * (1.0f + erff(u * 0.70710678118654752f)));
        } else {  // EPI_QKV
          const int sel = gn >> 10, cc = gn & 1023;
          const int h = cc >> 6, d = cc & 63;
          const int b = gm >> 10, n = gm & 1023;
          const int bh = b * HEADS + h;
          if (sel == 0)      Qo[((size_t)bh * SEQ + n) * HD + d] = (bf16)v;
          else if (sel == 1) Ko[((size_t)bh * SEQ + n) * HD + d] = (bf16)v;
          else               Vo[((size_t)bh * HD + d) * SEQ + n] = (bf16)v;
        }
      }
    }
  }
}

// ---------------------------------------------------------------------------
// Flash attention: grid (SEQ/128, B*H); 8 waves x 16 q-rows; 32-key chunks,
// double-buffered async K/V staging; row-sum via ones-WMMA.
// Q,K: [B,H,N,64] bf16;  Vt: [B,H,64,N] bf16;  O: [B,N,DIM] bf16
// ---------------------------------------------------------------------------
__global__ __launch_bounds__(256) void attn_kernel(
    const bf16* __restrict__ Q, const bf16* __restrict__ Kb,
    const bf16* __restrict__ Vt, bf16* __restrict__ O) {
  __shared__ __align__(16) bf16 Ks[2][32][HD + 8];     // keys x d, stride 72
  __shared__ __align__(16) bf16 Vs[2][HD][32 + 8];     // d x keys, stride 40
  __shared__ __align__(16) bf16 Ps[8][16][32 + 8];     // per-wave P tile
  const int bh = blockIdx.y;
  const int q0 = blockIdx.x * 128;
  const bf16* Qh = Q + (size_t)bh * SEQ * HD;
  const bf16* Kh = Kb + (size_t)bh * SEQ * HD;
  const bf16* Vh = Vt + (size_t)bh * HD * SEQ;
  const int t = threadIdx.x, w = t >> 5, lane = t & 31;
  const int mlane = lane & 15, half = lane >> 4;

  const bf16* qrow = Qh + (size_t)(q0 + w * 16 + mlane) * HD;
  const v16bf qa0 = frag_a(qrow, half);        // d = 0..31
  const v16bf qa1 = frag_a(qrow + 32, half);   // d = 32..63

  FragU ou;
#pragma unroll
  for (int e = 0; e < 8; e++) { ou.h[0][e] = (bf16)1.0f; ou.h[1][e] = (bf16)1.0f; }
  const v16bf ones = ou.v;                     // 32x16 all-ones B fragment

  v8f oacc[4] = {};
  float mrow[8], lrow[8];
#pragma unroll
  for (int r = 0; r < 8; r++) { mrow[r] = -3.0e38f; lrow[r] = 0.f; }

  const int krow = t >> 3, kch = (t & 7) * 8;
  const int vrow = t >> 2, vch = (t & 3) * 8;
  const unsigned int KBUF = (unsigned int)sizeof(Ks[0]);
  const unsigned int VBUF = (unsigned int)sizeof(Vs[0]);
  const unsigned int kA = lds_addr_of(&Ks[0][krow][kch]);
  const unsigned int vA = lds_addr_of(&Vs[0][vrow][vch]);

  // preload chunk 0 into buffer 0
  async_b128(kA, &Kh[(size_t)krow * HD + kch]);
  async_b128(vA, &Vh[(size_t)vrow * SEQ + vch]);

  int buf = 0;
  for (int j0 = 0; j0 < SEQ; j0 += 32) {
    wait_async0();
    __syncthreads();
    if (j0 + 32 < SEQ) {
      const unsigned int nb = (unsigned int)(buf ^ 1);
      async_b128(kA + nb * KBUF, &Kh[(size_t)(j0 + 32 + krow) * HD + kch]);
      async_b128(vA + nb * VBUF, &Vh[(size_t)vrow * SEQ + j0 + 32 + vch]);
    }

    v8f s[2];
#pragma unroll
    for (int tt = 0; tt < 2; tt++) {
      v8f z = {};
      const v16bf kb0 = frag_b(&Ks[buf][tt * 16 + mlane][0], half);   // d 0..31
      const v16bf kb1 = frag_b(&Ks[buf][tt * 16 + mlane][32], half);  // d 32..63
      z = WMMA_BF16(qa0, kb0, z);
      z = WMMA_BF16(qa1, kb1, z);
      s[tt] = z;
    }

#pragma unroll
    for (int r = 0; r < 8; r++) {
      const float s0 = s[0][r] * ATT_SCALE;
      const float s1 = s[1][r] * ATT_SCALE;
      float mx = fmaxf(s0, s1);
#pragma unroll
      for (int off = 1; off < 16; off <<= 1)
        mx = fmaxf(mx, __shfl_xor(mx, off, 32));
      const float mn = fmaxf(mrow[r], mx);
      const float alpha = __expf(mrow[r] - mn);
      mrow[r] = mn;
      const float p0 = __expf(s0 - mn);
      const float p1 = __expf(s1 - mn);
      lrow[r] *= alpha;
#pragma unroll
      for (int tt = 0; tt < 4; tt++) oacc[tt][r] *= alpha;
      Ps[w][half * 8 + r][mlane] = (bf16)p0;
      Ps[w][half * 8 + r][16 + mlane] = (bf16)p1;
    }
    asm volatile("s_wait_dscnt 0" ::: "memory");  // same-wave LDS write->read
    const v16bf pa = frag_a(&Ps[w][mlane][0], half);

    // row-sum of P via matrix unit: P(16x32) @ ones(32x16) -> rowsum in every col
    v8f ls = {};
    ls = WMMA_BF16(pa, ones, ls);
#pragma unroll
    for (int r = 0; r < 8; r++) lrow[r] += ls[r];

#pragma unroll
    for (int tt = 0; tt < 4; tt++) {
      const v16bf vb = frag_b(&Vs[buf][tt * 16 + mlane][0], half);
      oacc[tt] = WMMA_BF16(pa, vb, oacc[tt]);
    }
    buf ^= 1;
  }

  const int b = bh >> 4, h = bh & 15;
#pragma unroll
  for (int r = 0; r < 8; r++) {
    const float inv = 1.0f / lrow[r];
    const int grow = q0 + w * 16 + half * 8 + r;
#pragma unroll
    for (int tt = 0; tt < 4; tt++)
      O[((size_t)(b * SEQ + grow)) * DIMN + h * HD + tt * 16 + mlane] =
          (bf16)(oacc[tt][r] * inv);
  }
}

// ---------------------------------------------------------------------------
extern "C" void kernel_launch(void* const* d_in, const int* in_sizes, int n_in,
                              void* d_out, int out_size, void* d_ws,
                              size_t ws_size, hipStream_t stream) {
  (void)in_sizes; (void)n_in; (void)out_size; (void)ws_size;
  const float* x     = (const float*)d_in[0];
  const float* ln1_g = (const float*)d_in[1];
  const float* ln1_b = (const float*)d_in[2];
  const float* w_qkv = (const float*)d_in[3];
  const float* w_out = (const float*)d_in[4];
  const float* b_out = (const float*)d_in[5];
  const float* ln2_g = (const float*)d_in[6];
  const float* ln2_b = (const float*)d_in[7];
  const float* w1    = (const float*)d_in[8];
  const float* b1    = (const float*)d_in[9];
  const float* w2    = (const float*)d_in[10];
  const float* b2    = (const float*)d_in[11];
  float* X = (float*)d_out;  // x carried in-place across layers

  char* ws = (char*)d_ws;
  size_t off = 0;
  auto alloc = [&](size_t bytes) -> char* {
    char* p = ws + off;
    off += (bytes + 255) & ~(size_t)255;
    return p;
  };
  bf16* wqkvT = (bf16*)alloc((size_t)3072 * 1024 * 2);
  bf16* woutT = (bf16*)alloc((size_t)1024 * 1024 * 2);
  bf16* w1T   = (bf16*)alloc((size_t)4096 * 1024 * 2);
  bf16* w2T   = (bf16*)alloc((size_t)1024 * 4096 * 2);
  bf16* hbf   = (bf16*)alloc((size_t)ROWS * DIMN * 2);
  bf16* qbf   = (bf16*)alloc((size_t)ROWS * DIMN * 2);
  bf16* kbf   = (bf16*)alloc((size_t)ROWS * DIMN * 2);
  bf16* vbf   = (bf16*)alloc((size_t)ROWS * DIMN * 2);
  bf16* obf   = (bf16*)alloc((size_t)ROWS * DIMN * 2);
  bf16* gbf   = (bf16*)alloc((size_t)ROWS * MLPD * 2);

  hipMemcpyAsync(X, x, (size_t)ROWS * DIMN * sizeof(float),
                 hipMemcpyDeviceToDevice, stream);

  for (int l = 0; l < NLAYER; l++) {
    transpose_bf16_kernel<<<dim3(3072 / 32, 1024 / 32), 256, 0, stream>>>(
        w_qkv + (size_t)l * 1024 * 3072, wqkvT, 1024, 3072);
    transpose_bf16_kernel<<<dim3(1024 / 32, 1024 / 32), 256, 0, stream>>>(
        w_out + (size_t)l * 1024 * 1024, woutT, 1024, 1024);
    transpose_bf16_kernel<<<dim3(4096 / 32, 1024 / 32), 256, 0, stream>>>(
        w1 + (size_t)l * 1024 * 4096, w1T, 1024, 4096);
    transpose_bf16_kernel<<<dim3(1024 / 32, 4096 / 32), 256, 0, stream>>>(
        w2 + (size_t)l * 4096 * 1024, w2T, 4096, 1024);

    ln_kernel<<<ROWS, 256, 0, stream>>>(X, ln1_g + l * DIMN, ln1_b + l * DIMN, hbf);
    gemm_kernel<EPI_QKV><<<dim3(3072 / BN, ROWS / BM), 256, 0, stream>>>(
        hbf, wqkvT, nullptr, nullptr, nullptr, qbf, kbf, vbf, 3072, 1024);
    attn_kernel<<<dim3(SEQ / 128, BATCH * HEADS), 256, 0, stream>>>(
        qbf, kbf, vbf, obf);
    gemm_kernel<EPI_RES><<<dim3(1024 / BN, ROWS / BM), 256, 0, stream>>>(
        obf, woutT, b_out + l * DIMN, X, nullptr, nullptr, nullptr, nullptr,
        1024, 1024);

    ln_kernel<<<ROWS, 256, 0, stream>>>(X, ln2_g + l * DIMN, ln2_b + l * DIMN, hbf);
    gemm_kernel<EPI_GELU><<<dim3(4096 / BN, ROWS / BM), 256, 0, stream>>>(
        hbf, w1T, b1 + (size_t)l * MLPD, nullptr, gbf, nullptr, nullptr,
        nullptr, 4096, 1024);
    gemm_kernel<EPI_RES><<<dim3(1024 / BN, ROWS / BM), 256, 0, stream>>>(
        gbf, w2T, b2 + l * DIMN, X, nullptr, nullptr, nullptr, nullptr,
        1024, 4096);
  }
}
